// GraphNets_15745350107783
// MI455X (gfx1250) — matrix operand order repaired and checked by reading, not compiled, not run on hardware
//
#include <hip/hip_runtime.h>
#include <hip/hip_bf16.h>

// ---------------------------------------------------------------------------
// GraphNets forward for MI455X (gfx1250, wave32, WMMA 16x16x32 f16).
// GEMMs: [M x (NSEG*128)] @ [(NSEG*128) x 128], fused gather/concat on A,
// fused bias, fused BN column statistics in the epilogue (LDS ds_add_f32 +
// global atomic), and fused BN-normalize+SELU applied to A fragments on load
// (BNA template flag, per-column scale/shift cached in LDS, branchless SELU).
// All activations live in f16; only BN stats and d_out are f32.
// ---------------------------------------------------------------------------

typedef _Float16 half_t;
typedef _Float16 h16 __attribute__((ext_vector_type(16)));
typedef _Float16 h8  __attribute__((ext_vector_type(8)));
typedef _Float16 h4  __attribute__((ext_vector_type(4)));
typedef float    f8  __attribute__((ext_vector_type(8)));
typedef float    f4  __attribute__((ext_vector_type(4)));

#define HD 128
#define BN_EPS 1e-5f
#define SELU_S  1.0507009873554805f
#define SELU_SA 1.7580993408473766f   // scale * alpha

struct Seg { const half_t* src; const int* idx; };

// Branchless SELU: for t>0 the exp term is exp(0)-1 = 0; for t<=0 the linear
// term is 0.  Pure v_max/v_min/v_exp/v_fma -- no EXEC manipulation.
__device__ __forceinline__ float selu_f(float t) {
    float p = fmaxf(t, 0.f);
    float q = fminf(t, 0.f);
    return fmaf(SELU_S, p, SELU_SA * (__expf(q) - 1.f));
}

// ---------------------------------------------------------------------------
// Pack f32 weight [K x 128] into WMMA B-fragment order:
// packed[(((kt*8)+nt)*32 + lane)*16 + j] = W[kt*32 + (j>>3)*16 + (lane>>4)*8 + (j&7)][nt*16 + (lane&15)]
// so each lane's v16h B fragment is one contiguous 32-byte load.
// ---------------------------------------------------------------------------
__global__ void k_pack(const float* __restrict__ W, half_t* __restrict__ out, int K) {
    int total = K * HD;
    for (int idx = blockIdx.x * blockDim.x + threadIdx.x; idx < total;
         idx += gridDim.x * blockDim.x) {
        int j    = idx & 15;
        int lane = (idx >> 4) & 31;
        int nt   = (idx >> 9) & 7;
        int kt   = idx >> 12;
        int kk   = kt * 32 + (j >> 3) * 16 + (lane >> 4) * 8 + (j & 7);
        int nn   = nt * 16 + (lane & 15);
        out[idx] = (half_t)W[(size_t)kk * HD + nn];
    }
}

__global__ void k_cvt4(const float* __restrict__ in, half_t* __restrict__ out, long long n4) {
    for (long long i = blockIdx.x * (long long)blockDim.x + threadIdx.x; i < n4;
         i += (long long)gridDim.x * blockDim.x) {
        f4 v = ((const f4*)in)[i];
        h4 o;
#pragma unroll
        for (int j = 0; j < 4; ++j) o[j] = (half_t)v[j];
        ((h4*)out)[i] = o;
    }
}

__global__ void k_zero4(float* __restrict__ p, long long n4) {
    f4 z = {0.f, 0.f, 0.f, 0.f};
    for (long long i = blockIdx.x * (long long)blockDim.x + threadIdx.x; i < n4;
         i += (long long)gridDim.x * blockDim.x)
        ((f4*)p)[i] = z;
}

__global__ void k_ubr(const int* __restrict__ row, const int* __restrict__ batch,
                      int* __restrict__ ubr, int E) {
    int e = blockIdx.x * blockDim.x + threadIdx.x;
    if (e < E) ubr[e] = batch[row[e]];
}

__global__ void k_count(const int* __restrict__ idx, float* __restrict__ cnt, int n) {
    int i = blockIdx.x * blockDim.x + threadIdx.x;
    if (i < n) unsafeAtomicAdd(&cnt[idx[i]], 1.f);
}

// Column sums -> per-column BN scale/shift:  bn[c] = gamma*rstd,
// bn[128+c] = beta - mu*gamma*rstd.  (biased var, matches reference)
__global__ void k_bnprep(const float* __restrict__ stats, const float* __restrict__ gamma,
                         const float* __restrict__ beta, float* __restrict__ bn, float invM) {
    int c = threadIdx.x;   // 128 threads
    float mu   = stats[c] * invM;
    float var  = stats[c + 128] * invM - mu * mu;
    float sc   = gamma[c] * __frsqrt_rn(var + BN_EPS);
    bn[c]       = sc;
    bn[128 + c] = beta[c] - mu * sc;
}

// ---------------------------------------------------------------------------
// WMMA GEMM
// ---------------------------------------------------------------------------
template <int NSEG, bool BNA>
__global__ __launch_bounds__(256) void k_gemm(
    Seg s0, Seg s1, Seg s2, Seg s3, int M,
    const half_t* __restrict__ Bp, const float* __restrict__ bias,
    float* __restrict__ outF, half_t* __restrict__ outH,
    float* __restrict__ stats, const float* __restrict__ bnp)
{
    __shared__ alignas(16) float sred[256];
    __shared__ alignas(16) float sbn[256];
    const int tid  = threadIdx.x;
    const int wave = tid >> 5;
    const int lane = tid & 31;
    const int rowBase = blockIdx.x * 128 + wave * 16;
    const int m   = lane & 15;
    const int hi  = lane >> 4;        // 0 or 1 (lane half)
    const int kh  = hi * 8;           // A: K sub-offset per lane half
    const int row = rowBase + m;
    const int rowc = (row < M) ? row : (M - 1);   // clamp (EXEC stays all-ones)

    sred[tid] = 0.f;
    if (BNA) sbn[tid] = bnp[tid];
    __syncthreads();

    f8 acc[8];
    f8 zero = {0.f, 0.f, 0.f, 0.f, 0.f, 0.f, 0.f, 0.f};
#pragma unroll
    for (int i = 0; i < 8; ++i) acc[i] = zero;

    // Resolve all gather row pointers up-front so the index loads overlap.
    const Seg segarr[4] = { s0, s1, s2, s3 };
    const half_t* arowp[NSEG];
#pragma unroll
    for (int s = 0; s < NSEG; ++s) {
        const Seg sg = segarr[s];
        const int r = sg.idx ? sg.idx[rowc] : rowc;           // fused gather
        arowp[s] = sg.src + (size_t)r * HD;
    }

#pragma unroll
    for (int s = 0; s < NSEG; ++s) {
        const half_t* arow = arowp[s];
        const half_t* bseg = Bp + (size_t)s * (4 * 8 * 32 * 16);
#pragma unroll
        for (int kt = 0; kt < 4; ++kt) {          // 128 / 32 K-steps per segment
            // A fragment (16-bit 16x32 layout): halves 0..7 -> K=kh..kh+7,
            // halves 8..15 -> K=16+kh..16+kh+7
            h8 lo = *(const h8*)(arow + kt * 32 + kh);
            h8 hh = *(const h8*)(arow + kt * 32 + 16 + kh);
            if (BNA) {   // fused BatchNorm + SELU on the fly (prev layer's h)
                const int cb = kt * 32 + kh;      // multiple of 8 -> f4 aligned
                float sc[16], sh[16];
                *(f4*)&sc[0]  = *(const f4*)&sbn[cb];
                *(f4*)&sc[4]  = *(const f4*)&sbn[cb + 4];
                *(f4*)&sc[8]  = *(const f4*)&sbn[cb + 16];
                *(f4*)&sc[12] = *(const f4*)&sbn[cb + 20];
                *(f4*)&sh[0]  = *(const f4*)&sbn[128 + cb];
                *(f4*)&sh[4]  = *(const f4*)&sbn[128 + cb + 4];
                *(f4*)&sh[8]  = *(const f4*)&sbn[128 + cb + 16];
                *(f4*)&sh[12] = *(const f4*)&sbn[128 + cb + 20];
#pragma unroll
                for (int j = 0; j < 8; ++j) {
                    lo[j] = (half_t)selu_f(fmaf((float)lo[j], sc[j],     sh[j]));
                    hh[j] = (half_t)selu_f(fmaf((float)hh[j], sc[8 + j], sh[8 + j]));
                }
            }
            h16 a;
#pragma unroll
            for (int j = 0; j < 8; ++j) { a[j] = lo[j]; a[j + 8] = hh[j]; }

            const half_t* bkt = bseg + (size_t)kt * (8 * 32 * 16) + lane * 16;
            // Load all 8 B fragments first (one clause) ...
            h16 bfr[8];
#pragma unroll
            for (int nt = 0; nt < 8; ++nt)
                bfr[nt] = *(const h16*)(bkt + nt * (32 * 16));
            // ... then 8 back-to-back WMMAs.
#pragma unroll
            for (int nt = 0; nt < 8; ++nt)
                acc[nt] = __builtin_amdgcn_wmma_f32_16x16x32_f16(
                    false, a, false, bfr[nt], (short)0, acc[nt], false, false);
        }
    }

    // C/D layout: col = lane&15, rows = 8*(lane>>4) + i for VGPR i
    const int n  = lane & 15;
    const int mb = hi * 8;
#pragma unroll
    for (int nt = 0; nt < 8; ++nt) {
        const int c = nt * 16 + n;
        const float bc = bias[c];
        float sv = 0.f, sq = 0.f;
#pragma unroll
        for (int i = 0; i < 8; ++i) {
            const int r = rowBase + mb + i;
            const float v = acc[nt][i] + bc;
            if (r < M) {
                if (outF) outF[(size_t)r * HD + c] = v;
                if (outH) outH[(size_t)r * HD + c] = (half_t)v;
                sv += v; sq += v * v;
            }
        }
        if (stats) {
            atomicAdd(&sred[c], sv);          // ds_add_f32
            atomicAdd(&sred[c + 128], sq);
        }
    }
    if (stats) {
        __syncthreads();
        unsafeAtomicAdd(&stats[tid], sred[tid]);   // global_atomic_add_f32
    }
}

// ---------------------------------------------------------------------------
// Segment mean: atomic accumulate (f16 source) then finalize to f16.
// ---------------------------------------------------------------------------
__global__ void k_seg_accum4(const half_t* __restrict__ vals, const int* __restrict__ idx,
                             float* __restrict__ sum, long long n4)
{
    for (long long i = blockIdx.x * (long long)blockDim.x + threadIdx.x; i < n4;
         i += (long long)gridDim.x * blockDim.x) {
        long long r = i >> 5;                  // (i*4)/128
        int c0 = (int)((i << 2) & (HD - 1));
        int t = idx[r];
        h4 v = ((const h4*)vals)[i];
        float* dst = sum + (size_t)t * HD + c0;
#pragma unroll
        for (int j = 0; j < 4; ++j) unsafeAtomicAdd(dst + j, (float)v[j]);
    }
}

__global__ void k_seg_fin4(const float* __restrict__ sum, const float* __restrict__ cnt,
                           half_t* __restrict__ out, long long n4)
{
    for (long long i = blockIdx.x * (long long)blockDim.x + threadIdx.x; i < n4;
         i += (long long)gridDim.x * blockDim.x) {
        long long r = i >> 5;                  // (i*4)/128
        float inv = 1.f / fmaxf(cnt[r], 1.f);
        f4 v = ((const f4*)sum)[i];
        h4 o;
#pragma unroll
        for (int j = 0; j < 4; ++j) o[j] = (half_t)(v[j] * inv);
        ((h4*)out)[i] = o;
    }
}

// ---------------------------------------------------------------------------
// Host orchestration
// ---------------------------------------------------------------------------
static inline int cdiv(long long a, long long b) { return (int)((a + b - 1) / b); }

extern "C" void kernel_launch(void* const* d_in, const int* in_sizes, int n_in,
                              void* d_out, int out_size, void* d_ws, size_t ws_size,
                              hipStream_t stream)
{
    const int N = in_sizes[0] / HD;
    const int E = in_sizes[1] / 2;
    const int B = in_sizes[3] / HD;

    const float* x_f  = (const float*)d_in[0];
    const int*   eidx = (const int*)d_in[1];
    const float* ea_f = (const float*)d_in[2];
    const float* u_f  = (const float*)d_in[3];
    const int*   batch = (const int*)d_in[4];
    const int*   row = eidx;
    const int*   col = eidx + E;

    // params: 3 layers x {edge, node1, node2, global} x
    //         {W0,b0,g0,be0, W1,b1,g1,be1, Wf,bf}  (setup_inputs insertion order)
    auto PP = [&](int L, int m, int i) -> const float* {
        return (const float*)d_in[5 + L * 40 + m * 10 + i];
    };
    const int K0s[4] = { 512, 256, 384, 256 };  // edge, node1, node2, global

    // bump allocator over d_ws
    char* base = (char*)d_ws;
    size_t off = 0;
    auto alloc = [&](size_t bytes) -> void* {
        void* p = base + off;
        off += (bytes + 255) & ~(size_t)255;
        return p;
    };

    half_t* rawA   = (half_t*)alloc((size_t)E * HD * 2); // pre-BN h of hidden layer 1
    half_t* rawB   = (half_t*)alloc((size_t)E * HD * 2); // pre-BN h of hidden layer 2
    half_t* Eh     = (half_t*)alloc((size_t)E * HD * 2); // edge_attr (f16)
    half_t* Xh     = (half_t*)alloc((size_t)N * HD * 2); // x (f16)
    half_t* Uh     = (half_t*)alloc((size_t)B * HD * 2); // u (f16)
    half_t* Aggh   = (half_t*)alloc((size_t)N * HD * 2); // scatter-mean of messages
    half_t* Gh     = (half_t*)alloc((size_t)B * HD * 2); // scatter-mean of x by graph
    float*  aggsum = (float*)alloc((size_t)N * HD * 4);
    float*  aggcnt = (float*)alloc((size_t)N * 4);
    float*  gsum   = (float*)alloc((size_t)B * HD * 4);
    float*  gcnt   = (float*)alloc((size_t)B * 4);
    int*    ubr    = (int*)alloc((size_t)E * 4);
    float*  stats  = (float*)alloc(256 * 4);
    float*  bnbuf  = (float*)alloc(256 * 4);

    half_t* pk[3][4][3];
    for (int L = 0; L < 3; ++L)
        for (int m = 0; m < 4; ++m) {
            pk[L][m][0] = (half_t*)alloc((size_t)K0s[m] * HD * 2);
            pk[L][m][1] = (half_t*)alloc((size_t)HD * HD * 2);
            pk[L][m][2] = (half_t*)alloc((size_t)HD * HD * 2);
        }

    // ---- pack all weights to f16 fragment layout ----
    for (int L = 0; L < 3; ++L)
        for (int m = 0; m < 4; ++m) {
            int K0 = K0s[m];
            k_pack<<<cdiv((long long)K0 * HD, 256), 256, 0, stream>>>(PP(L, m, 0), pk[L][m][0], K0);
            k_pack<<<cdiv((long long)HD * HD, 256), 256, 0, stream>>>(PP(L, m, 4), pk[L][m][1], HD);
            k_pack<<<cdiv((long long)HD * HD, 256), 256, 0, stream>>>(PP(L, m, 8), pk[L][m][2], HD);
        }

    // ---- initial conversions, gather index, layer-invariant counts ----
    k_cvt4<<<cdiv((long long)N * HD / 4, 256), 256, 0, stream>>>(x_f, Xh, (long long)N * HD / 4);
    k_cvt4<<<cdiv((long long)E * HD / 4, 256), 256, 0, stream>>>(ea_f, Eh, (long long)E * HD / 4);
    k_cvt4<<<cdiv((long long)B * HD / 4, 256), 256, 0, stream>>>(u_f, Uh, (long long)B * HD / 4);
    k_ubr<<<cdiv(E, 256), 256, 0, stream>>>(row, batch, ubr, E);
    k_zero4<<<cdiv((long long)N / 4, 256), 256, 0, stream>>>(aggcnt, (long long)N / 4);
    k_zero4<<<1, 64, 0, stream>>>(gcnt, (long long)B / 4);
    k_count<<<cdiv(E, 256), 256, 0, stream>>>(col, aggcnt, E);
    k_count<<<cdiv(N, 256), 256, 0, stream>>>(batch, gcnt, N);

    Seg snull{nullptr, nullptr};

    auto launch_gemm = [&](int nseg, Seg a, Seg b, Seg c, Seg d, int M,
                           const half_t* Bp, const float* bias,
                           float* oF, half_t* oH, float* st, const float* bnp) {
        const int g = (M + 127) / 128;
        if (nseg == 1) {
            if (bnp) k_gemm<1, true ><<<g, 256, 0, stream>>>(a, b, c, d, M, Bp, bias, oF, oH, st, bnp);
            else     k_gemm<1, false><<<g, 256, 0, stream>>>(a, b, c, d, M, Bp, bias, oF, oH, st, nullptr);
        } else if (nseg == 2) {
            k_gemm<2, false><<<g, 256, 0, stream>>>(a, b, c, d, M, Bp, bias, oF, oH, st, nullptr);
        } else if (nseg == 3) {
            k_gemm<3, false><<<g, 256, 0, stream>>>(a, b, c, d, M, Bp, bias, oF, oH, st, nullptr);
        } else {
            k_gemm<4, false><<<g, 256, 0, stream>>>(a, b, c, d, M, Bp, bias, oF, oH, st, nullptr);
        }
    };

    // MLP: GEMM1(+stats) -> bnprep -> GEMM2(BNA,+stats) -> bnprep -> GEMM3(BNA)
    auto run_mlp = [&](Seg s0, Seg s1, Seg s2, Seg s3, int nseg, int M,
                       int L, int m, float* outF, half_t* outH) {
        const float invM = 1.f / (float)M;
        k_zero4<<<1, 64, 0, stream>>>(stats, 64);
        launch_gemm(nseg, s0, s1, s2, s3, M, pk[L][m][0], PP(L, m, 1),
                    nullptr, rawA, stats, nullptr);
        k_bnprep<<<1, 128, 0, stream>>>(stats, PP(L, m, 2), PP(L, m, 3), bnbuf, invM);

        k_zero4<<<1, 64, 0, stream>>>(stats, 64);
        launch_gemm(1, Seg{rawA, nullptr}, snull, snull, snull, M, pk[L][m][1], PP(L, m, 5),
                    nullptr, rawB, stats, bnbuf);
        k_bnprep<<<1, 128, 0, stream>>>(stats, PP(L, m, 6), PP(L, m, 7), bnbuf, invM);

        launch_gemm(1, Seg{rawB, nullptr}, snull, snull, snull, M, pk[L][m][2], PP(L, m, 9),
                    outF, outH, nullptr, bnbuf);
    };

    for (int L = 0; L < 3; ++L) {
        // ---- edge update: mlp(cat[x[row], x[col], edge_attr, u[batch[row]]]) -> Eh ----
        run_mlp(Seg{Xh, row}, Seg{Xh, col}, Seg{Eh, nullptr}, Seg{Uh, ubr}, 4, E, L, 0,
                nullptr, Eh);

        // ---- node update part 1: m = mlp(cat[x[row], edge_attr']) -> rawA (f16) ----
        run_mlp(Seg{Xh, row}, Seg{Eh, nullptr}, snull, snull, 2, E, L, 1, nullptr, rawA);

        // agg = scatter_mean(m, col, N) -> Aggh
        k_zero4<<<cdiv((long long)N * HD / 4, 256), 256, 0, stream>>>(aggsum, (long long)N * HD / 4);
        k_seg_accum4<<<cdiv((long long)E * HD / 4, 256), 256, 0, stream>>>(rawA, col, aggsum,
                                                                           (long long)E * HD / 4);
        k_seg_fin4<<<cdiv((long long)N * HD / 4, 256), 256, 0, stream>>>(aggsum, aggcnt, Aggh,
                                                                         (long long)N * HD / 4);

        // ---- node update part 2: x' = mlp(cat[x, agg, u[batch]]) -> Xh (in place) ----
        run_mlp(Seg{Xh, nullptr}, Seg{Aggh, nullptr}, Seg{Uh, batch}, snull, 3, N, L, 2,
                nullptr, Xh);

        // g = scatter_mean(x', batch, B) -> Gh
        k_zero4<<<cdiv((long long)B * HD / 4, 256), 256, 0, stream>>>(gsum, (long long)B * HD / 4);
        k_seg_accum4<<<cdiv((long long)N * HD / 4, 256), 256, 0, stream>>>(Xh, batch, gsum,
                                                                           (long long)N * HD / 4);
        k_seg_fin4<<<cdiv((long long)B * HD / 4, 256), 256, 0, stream>>>(gsum, gcnt, Gh,
                                                                         (long long)B * HD / 4);

        // ---- global update: u' = mlp(cat[u, g]) ----
        if (L == 2)
            run_mlp(Seg{Uh, nullptr}, Seg{Gh, nullptr}, snull, snull, 2, B, L, 3,
                    (float*)d_out, nullptr);
        else
            run_mlp(Seg{Uh, nullptr}, Seg{Gh, nullptr}, snull, snull, 2, B, L, 3,
                    nullptr, Uh);
    }
}
